// Attention_71098888618623
// MI455X (gfx1250) — matrix operand order repaired and checked by reading, not compiled, not run on hardware
//
#include <hip/hip_runtime.h>
#include <hip/hip_bf16.h>
#include <math.h>
#include <stdint.h>

// Problem constants (from reference)
static constexpr int CB  = 32;    // batch
static constexpr int CS  = 512;   // sequence (patches)
static constexpr int CD  = 768;   // hidden dim
static constexpr int CH  = 12;    // heads
static constexpr int CK  = 8;     // topic groups
static constexpr int CHD = 64;    // head dim

typedef __attribute__((ext_vector_type(16))) _Float16 v16h;
typedef __attribute__((ext_vector_type(8)))  _Float16 v8h;
typedef __attribute__((ext_vector_type(8)))  float    v8f;

union V16U { v16h v; v8h h[2]; };

// ---------------- WMMA fragment helpers (wave32, 16x16x32 f16) ----------------
// A (16x32, MxK) per ISA: lane L (&15)=row M; K chunk0 = 8*(L>>4)..+8, chunk1 = 16+8*(L>>4)..+8
__device__ __forceinline__ v16h load_a_frag(const _Float16* p, int ld) {
  const int lane = threadIdx.x & 31;
  const int hi   = lane >> 4;
  const _Float16* q = p + (size_t)(lane & 15) * ld + hi * 8;
  V16U u;
  u.h[0] = *(const v8h*)q;
  u.h[1] = *(const v8h*)(q + 16);
  return u.v;
}

// B (32x16, KxN) for A*W^T: lane L -> column N = L&15 = row of W; K chunk0 =
// 8*(L>>4)..+8, chunk1 = 16+8*(L>>4)..+8 (same K order as A).
__device__ __forceinline__ v16h load_b_frag(const _Float16* p, int ld) {
  const int lane = threadIdx.x & 31;
  const _Float16* q = p + (size_t)(lane & 15) * ld + (lane >> 4) * 8;
  V16U u;
  u.h[0] = *(const v8h*)q;
  u.h[1] = *(const v8h*)(q + 16);
  return u.v;
}

// A-fragment from a 16-row f16 tile in LDS with row stride `ldh` halves
__device__ __forceinline__ v16h lds_a_frag(const _Float16* p, int ldh) {
  const int lane = threadIdx.x & 31;
  const int hi   = lane >> 4;
  const _Float16* q = p + (lane & 15) * ldh + hi * 8;
  V16U u;
  u.h[0] = *(const v8h*)q;
  u.h[1] = *(const v8h*)(q + 16);
  return u.v;
}

__device__ __forceinline__ v8f wmma16(v16h a, v16h b, v8f c) {
  return __builtin_amdgcn_wmma_f32_16x16x32_f16(false, a, false, b, (short)0, c,
                                                false, false);
}

// CDNA5 async copy: global -> LDS, 16 bytes per lane, tracked by ASYNCcnt.
__device__ __forceinline__ void async_g2l_b128(const _Float16* gaddr,
                                               _Float16* ldsaddr) {
  const uint32_t lo = (uint32_t)(uintptr_t)ldsaddr;  // DS byte offset
  asm volatile("global_load_async_to_lds_b128 %0, %1, off"
               :: "v"(lo), "v"(gaddr)
               : "memory");
}

__device__ __forceinline__ void wait_async0() {
  asm volatile("s_wait_asynccnt 0x0" ::: "memory");
}

// ---------------- elementwise conversion ----------------
__global__ void cvt_f32_f16(const float* __restrict__ in, _Float16* __restrict__ out,
                            size_t n) {
  size_t i = (size_t)blockIdx.x * blockDim.x + threadIdx.x;
  if (i < n) out[i] = (_Float16)in[i];
}

// ---------------- async-pipelined WMMA GEMM: out = A[M,K]*W[N,K]^T + bias -----
// Block: 256 threads / 8 waves. Block tile 64(M) x 128(N); wave tile 64x16.
// A tile (64x32 f16) staged in LDS via global_load_async_to_lds_b128 (one b128
// per thread per K-step), double-buffered; B fragments prefetched one K-step
// ahead in registers. LDS row stride 40 halves (80 B = 20 banks) ->
// conflict-free ds_load_b128 across a 16-lane half.
static constexpr int ALD = 40;  // padded LDS row stride in halves

__global__ void __launch_bounds__(256)
gemm_nt(const _Float16* __restrict__ A, const _Float16* __restrict__ W,
        const float* __restrict__ bias, const float* __restrict__ resid,
        float* __restrict__ outF, _Float16* __restrict__ outH,
        int M, int Nn, int Kd) {
  __shared__ _Float16 atile[2][64 * ALD];  // 2 x 5120 B

  const int tidb = threadIdx.x;
  const int wave = tidb >> 5;
  const int lane = tidb & 31;
  const int hi   = lane >> 4;
  const int lo   = lane & 15;
  const int m0 = blockIdx.x * 64;
  const int n0 = blockIdx.y * 128 + wave * 16;

  // per-thread async staging slot: row = tid>>2 (0..63), col chunk = tid&3
  const int trow = tidb >> 2;
  const int tcol = tidb & 3;
  const _Float16* agbase = A + (size_t)(m0 + trow) * Kd + tcol * 8;
  _Float16* albase = &atile[0][0] + trow * ALD + tcol * 8;

  // prologue: stage k=0 tile, prefetch first B fragment
  async_g2l_b128(agbase, albase);
  v16h bf = load_b_frag(W + (size_t)n0 * Kd, Kd);
  wait_async0();
  __syncthreads();

  v8f acc[4] = {};
  for (int k = 0; k < Kd; k += 32) {
    const int cur = (k >> 5) & 1;
    const bool more = (k + 32) < Kd;
    if (more)  // overlap next copy with this step's WMMAs
      async_g2l_b128(agbase + k + 32, albase + (cur ^ 1) * (64 * ALD));
    v16h bnext = bf;
    if (more)
      bnext = load_b_frag(W + (size_t)n0 * Kd + k + 32, Kd);

#pragma unroll
    for (int i = 0; i < 4; ++i) {
      v16h af = lds_a_frag(&atile[cur][i * 16 * ALD], ALD);
      acc[i] = wmma16(af, bf, acc[i]);
    }
    bf = bnext;
    wait_async0();    // own async transfers done
    __syncthreads();  // everyone's transfers done; safe to read/overwrite
  }

  const int nc = n0 + lo;
  const float bv = bias ? bias[nc] : 0.f;
#pragma unroll
  for (int i = 0; i < 4; ++i) {
#pragma unroll
    for (int r = 0; r < 8; ++r) {
      const int mr = m0 + 16 * i + r + 8 * hi;
      float v = acc[i][r] + bv;
      if (resid) v += resid[(size_t)mr * Nn + nc];
      if (outF) outF[(size_t)mr * Nn + nc] = v;
      else      outH[(size_t)mr * Nn + nc] = (_Float16)v;
    }
  }
}

// ---------------- segment mean over batch by group id ----------------
__global__ void seg_mean(const float* __restrict__ y, const int* __restrict__ tid,
                         float* __restrict__ means) {
  const size_t nd = (size_t)blockIdx.x * blockDim.x + threadIdx.x;
  const size_t ND = (size_t)CS * CD;
  if (nd >= ND) return;
  float s[CK]; int c[CK];
#pragma unroll
  for (int g = 0; g < CK; ++g) { s[g] = 0.f; c[g] = 0; }
  for (int b = 0; b < CB; ++b) {
    int g = tid[b] - 1;
    g = (g < 0) ? 0 : (g >= CK ? CK - 1 : g);
    const float v = y[(size_t)b * ND + nd];
#pragma unroll
    for (int gi = 0; gi < CK; ++gi)
      if (g == gi) { s[gi] += v; c[gi] += 1; }
  }
#pragma unroll
  for (int g = 0; g < CK; ++g)
    means[(size_t)g * ND + nd] = s[g] / fmaxf((float)c[g], 1.0f);
}

// contexts = means[seg[b]] + x  (emitted directly as f16 for the T projection)
__global__ void make_ctx(const float* __restrict__ x, const float* __restrict__ means,
                         const int* __restrict__ tid, _Float16* __restrict__ ch) {
  const size_t i = (size_t)blockIdx.x * blockDim.x + threadIdx.x;
  const size_t ND = (size_t)CS * CD;
  const size_t total = (size_t)CB * ND;
  if (i >= total) return;
  const int b = (int)(i / ND);
  const size_t nd = i % ND;
  int g = tid[b] - 1;
  g = (g < 0) ? 0 : (g >= CK ? CK - 1 : g);
  ch[i] = (_Float16)(x[i] + means[(size_t)g * ND + nd]);
}

// V [B,S,D] f16 -> VT [B*H, HD, S] f16 (contiguous keys per hd row)
__global__ void transpose_heads(const _Float16* __restrict__ Vf,
                                _Float16* __restrict__ VT) {
  const size_t i = (size_t)blockIdx.x * blockDim.x + threadIdx.x;
  const size_t total = (size_t)CB * CH * CHD * CS;
  if (i >= total) return;
  const int n  = (int)(i % CS);
  const int hd = (int)((i / CS) % CHD);
  const int h  = (int)((i / ((size_t)CS * CHD)) % CH);
  const int b  = (int)(i / ((size_t)CS * CHD * CH));
  VT[i] = Vf[((size_t)b * CS + n) * CD + h * CHD + hd];
}

// ---------------- fused attention stage: O = softmax(Qr*Kr^T * scale) * V ------
// Qr, Kr flat [B*S, D] f16 (head slice indexed); VT [B*H, HD, S] f16.
// Oflat [B*S, D] f16; OT (optional) [B*H, HD, S] f16.
__global__ void __launch_bounds__(256)
attn_stage(const _Float16* __restrict__ Qr, const _Float16* __restrict__ Kr,
           const _Float16* __restrict__ VT, _Float16* __restrict__ Oflat,
           _Float16* __restrict__ OT, float scale) {
  __shared__ _Float16 plds[8 * 16 * 32];  // per-wave 16x32 P staging (8 KB)

  const int bh = blockIdx.x;
  const int b  = bh / CH;
  const int h  = bh % CH;
  const int wave = threadIdx.x >> 5;
  const int lane = threadIdx.x & 31;
  const int hi = lane >> 4;
  const int lo = lane & 15;
  const size_t rowbase = (size_t)b * CS;
  const int hoff = h * CHD;
  const _Float16* vtb = VT + (size_t)bh * CHD * CS;
  _Float16* wl = plds + wave * (16 * 32);

  for (int qg = 0; qg < 4; ++qg) {
    const int q0 = wave * 64 + qg * 16;  // 16 query rows for this pass
    const v16h a0 = load_a_frag(Qr + (rowbase + q0) * CD + hoff + 0,  CD);
    const v16h a1 = load_a_frag(Qr + (rowbase + q0) * CD + hoff + 32, CD);

    v8f acc[4] = {};
    float mrow[8], lrow[8];
#pragma unroll
    for (int r = 0; r < 8; ++r) { mrow[r] = -1e30f; lrow[r] = 0.f; }

    for (int kb = 0; kb < CS; kb += 32) {
      const _Float16* kp = Kr + (rowbase + kb) * CD + hoff;
      // S tiles: rows = queries, cols = keys kb..kb+31
      v8f s0 = {};
      s0 = wmma16(a0, load_b_frag(kp, CD), s0);
      s0 = wmma16(a1, load_b_frag(kp + 32, CD), s0);
      v8f s1 = {};
      s1 = wmma16(a0, load_b_frag(kp + (size_t)16 * CD, CD), s1);
      s1 = wmma16(a1, load_b_frag(kp + (size_t)16 * CD + 32, CD), s1);

#pragma unroll
      for (int r = 0; r < 8; ++r) { s0[r] *= scale; s1[r] *= scale; }

      // online softmax; each C-layout row lives in one VGPR across a 16-lane half
#pragma unroll
      for (int r = 0; r < 8; ++r) {
        float mx = fmaxf(s0[r], s1[r]);
        mx = fmaxf(mx, __shfl_xor(mx, 1, 16));
        mx = fmaxf(mx, __shfl_xor(mx, 2, 16));
        mx = fmaxf(mx, __shfl_xor(mx, 4, 16));
        mx = fmaxf(mx, __shfl_xor(mx, 8, 16));
        const float mnew = fmaxf(mrow[r], mx);
        const float f  = __expf(mrow[r] - mnew);
        const float p0 = __expf(s0[r] - mnew);
        const float p1 = __expf(s1[r] - mnew);
        float rs = p0 + p1;
        rs += __shfl_xor(rs, 1, 16);
        rs += __shfl_xor(rs, 2, 16);
        rs += __shfl_xor(rs, 4, 16);
        rs += __shfl_xor(rs, 8, 16);
        lrow[r] = lrow[r] * f + rs;
        mrow[r] = mnew;
        acc[0][r] *= f; acc[1][r] *= f; acc[2][r] *= f; acc[3][r] *= f;
        // stage P (f16) row-major into LDS for C-layout -> A-fragment relayout
        const int m = r + 8 * hi;
        wl[m * 32 + lo]      = (_Float16)p0;
        wl[m * 32 + 16 + lo] = (_Float16)p1;
      }

      const v16h ap = lds_a_frag(wl, 32);  // same-wave DS ops are in order
#pragma unroll
      for (int j = 0; j < 4; ++j) {
        const v16h bv = load_b_frag(vtb + (size_t)(j * 16) * CS + kb, CS);
        acc[j] = wmma16(ap, bv, acc[j]);
      }
    }

    // epilogue: normalize and store (flat head-concat layout + optional [hd,seq])
#pragma unroll
    for (int r = 0; r < 8; ++r) {
      const float inv = 1.0f / lrow[r];
      const int m = q0 + r + 8 * hi;
#pragma unroll
      for (int j = 0; j < 4; ++j) {
        const float o = acc[j][r] * inv;
        const int n = j * 16 + lo;
        Oflat[(rowbase + m) * CD + hoff + n] = (_Float16)o;
        if (OT) OT[(size_t)bh * CHD * CS + (size_t)n * CS + m] = (_Float16)o;
      }
    }
  }
}

// ------------------------------- host-side graph -------------------------------
extern "C" void kernel_launch(void* const* d_in, const int* in_sizes, int n_in,
                              void* d_out, int out_size, void* d_ws, size_t ws_size,
                              hipStream_t stream) {
  (void)in_sizes; (void)n_in; (void)out_size; (void)ws_size;

  const float* x   = (const float*)d_in[0];
  const int*   tid = (const int*)d_in[1];
  const float* Wq  = (const float*)d_in[2];
  const float* bq  = (const float*)d_in[3];
  const float* Wk  = (const float*)d_in[4];
  const float* bk  = (const float*)d_in[5];
  const float* Wv  = (const float*)d_in[6];
  const float* bv  = (const float*)d_in[7];
  const float* Wo  = (const float*)d_in[8];
  const float* bo  = (const float*)d_in[9];
  const float* Wc  = (const float*)d_in[10];
  const float* bc  = (const float*)d_in[11];
  float* out = (float*)d_out;

  const size_t M  = (size_t)CB * CS;      // 16384 rows
  const size_t MD = M * CD;               // activation elements
  const size_t WD = (size_t)CD * CD;      // weight elements

  char* w = (char*)d_ws;
  auto alloc = [&](size_t bytes) -> char* {
    char* p = w;
    w += (bytes + 255) & ~(size_t)255;
    return p;
  };
  _Float16* xh    = (_Float16*)alloc(MD * 2);
  _Float16* wqh   = (_Float16*)alloc(WD * 2);
  _Float16* wkh   = (_Float16*)alloc(WD * 2);
  _Float16* wvh   = (_Float16*)alloc(WD * 2);
  _Float16* woh   = (_Float16*)alloc(WD * 2);
  _Float16* wch   = (_Float16*)alloc(WD * 2);
  float*    y     = (float*)   alloc(MD * 4);
  float*    means = (float*)   alloc((size_t)CK * CS * CD * 4);
  _Float16* ch    = (_Float16*)alloc(MD * 2);
  _Float16* Qf    = (_Float16*)alloc(MD * 2);
  _Float16* Kf    = (_Float16*)alloc(MD * 2);
  _Float16* Vf    = (_Float16*)alloc(MD * 2);
  _Float16* Tf    = (_Float16*)alloc(MD * 2);
  _Float16* VTv   = (_Float16*)alloc(MD * 2);
  _Float16* Vt1   = (_Float16*)alloc(MD * 2);   // stage-1 flat output (unused later)
  _Float16* VtT   = (_Float16*)alloc(MD * 2);   // stage-1 transposed output
  _Float16* Ctx2  = (_Float16*)alloc(MD * 2);   // stage-2 output (head-concat)

  const int TPB = 256;
  const float scale = 0.125f;  // 1/sqrt(64)

  // 1) f32 -> f16 conversions
  cvt_f32_f16<<<dim3((unsigned)((MD + TPB - 1) / TPB)), TPB, 0, stream>>>(x, xh, MD);
  cvt_f32_f16<<<dim3((unsigned)((WD + TPB - 1) / TPB)), TPB, 0, stream>>>(Wq, wqh, WD);
  cvt_f32_f16<<<dim3((unsigned)((WD + TPB - 1) / TPB)), TPB, 0, stream>>>(Wk, wkh, WD);
  cvt_f32_f16<<<dim3((unsigned)((WD + TPB - 1) / TPB)), TPB, 0, stream>>>(Wv, wvh, WD);
  cvt_f32_f16<<<dim3((unsigned)((WD + TPB - 1) / TPB)), TPB, 0, stream>>>(Wo, woh, WD);
  cvt_f32_f16<<<dim3((unsigned)((WD + TPB - 1) / TPB)), TPB, 0, stream>>>(Wc, wch, WD);

  const dim3 ggrid((unsigned)(M / 64), (unsigned)(CD / 128));

  // 2) conv1x1: y = x + x*Wc^T + bc  (f32 for accurate segment means)
  gemm_nt<<<ggrid, TPB, 0, stream>>>(xh, wch, bc, x, y, nullptr,
                                     (int)M, CD, CD);

  // 3) segment mean over batch, then contexts = means[seg] + x (f16)
  seg_mean<<<dim3((unsigned)(((size_t)CS * CD + TPB - 1) / TPB)), TPB, 0, stream>>>(
      y, tid, means);
  make_ctx<<<dim3((unsigned)((MD + TPB - 1) / TPB)), TPB, 0, stream>>>(
      x, means, tid, ch);

  // 4) projections (f16 outputs)
  gemm_nt<<<ggrid, TPB, 0, stream>>>(xh, wqh, bq, nullptr, nullptr, Qf, (int)M, CD, CD);
  gemm_nt<<<ggrid, TPB, 0, stream>>>(xh, wkh, bk, nullptr, nullptr, Kf, (int)M, CD, CD);
  gemm_nt<<<ggrid, TPB, 0, stream>>>(xh, wvh, bv, nullptr, nullptr, Vf, (int)M, CD, CD);
  gemm_nt<<<ggrid, TPB, 0, stream>>>(ch, wqh, bq, nullptr, nullptr, Tf, (int)M, CD, CD);

  // 5) V -> [bh, hd, seq] for contiguous P*V B-fragments
  transpose_heads<<<dim3((unsigned)((MD + TPB - 1) / TPB)), TPB, 0, stream>>>(Vf, VTv);

  // 6) dual-stage attention (one block per (b,h))
  attn_stage<<<dim3(CB * CH), TPB, 0, stream>>>(Tf, Kf, VTv, Vt1, VtT, scale);
  attn_stage<<<dim3(CB * CH), TPB, 0, stream>>>(Qf, Tf, VtT, Ctx2, nullptr, scale);

  // 7) output projection -> d_out (f32)
  gemm_nt<<<ggrid, TPB, 0, stream>>>(Ctx2, woh, bo, nullptr, out, nullptr,
                                     (int)M, CD, CD);
}
